// FGN_layer_64647847739876
// MI455X (gfx1250) — matrix operand order repaired and checked by reading, not compiled, not run on hardware
//
#include <hip/hip_runtime.h>

// Problem dims
#define B_DIM   8192
#define IN_DIM  2048
#define OUT_DIM 2048

// Tiling: workgroup tile 128(M) x 256(N), K-step 32, 8 wave32 waves (2x4),
// each wave owns a 64x64 sub-tile -> 32 WMMAs per k-step per wave.
#define BM 128
#define BN 256
#define BK 32
#define LDS_ROW 80                       // 64B data + pad, multiple of 16
#define TILE_A_BYTES (128 * LDS_ROW)     // 10240
#define TILE_B_BYTES (256 * LDS_ROW)     // 20480
#define STAGE_BYTES  (TILE_A_BYTES + 2 * TILE_B_BYTES)   // 51200
#define SMEM_BYTES   (2 * STAGE_BYTES)                   // 102400

typedef __attribute__((ext_vector_type(16))) __bf16 v16bf;
typedef __attribute__((ext_vector_type(8)))  float  v8f;
typedef __attribute__((ext_vector_type(4)))  int    v4i;

// --- CDNA5 async global->LDS path (probe via __has_builtin) -----------------
#if defined(__has_builtin)
#  if __has_builtin(__builtin_amdgcn_global_load_async_to_lds_b128)
#    define HAVE_ASYNC 1
#  endif
#endif
#ifndef HAVE_ASYNC
#  define HAVE_ASYNC 0
#endif

#if defined(__has_builtin) && __has_builtin(__builtin_amdgcn_s_wait_asynccnt)
#  define WAIT_ASYNC() __builtin_amdgcn_s_wait_asynccnt(0)
#elif HAVE_ASYNC
#  define WAIT_ASYNC() asm volatile("s_wait_asynccnt 0x0" ::: "memory")
#else
#  define WAIT_ASYNC() do {} while (0)
#endif

union Frag {
    uint4  q[2];
    v16bf  v;
};

__device__ inline unsigned short f2bf(float f) {
    union { float f; unsigned u; } x;
    x.f = f;
    unsigned r = x.u + 0x7FFFu + ((x.u >> 16) & 1u);   // round-to-nearest-even
    return (unsigned short)(r >> 16);
}

// ---------------------------------------------------------------------------
// Pre-pass: convert W, C to bf16; bias[o] = -sum(W*C); csq[o] = sum(C*C);
// isig[o] = 1/sigs[o]^2. All reductions in fp32 on original data.
// ---------------------------------------------------------------------------
__global__ __launch_bounds__(256)
void prep_wc(const float* __restrict__ W, const float* __restrict__ C,
             const float* __restrict__ sigs,
             unsigned short* __restrict__ wb, unsigned short* __restrict__ cb,
             float* __restrict__ bias, float* __restrict__ csq,
             float* __restrict__ isig)
{
    __shared__ float r1[256];
    __shared__ float r2[256];
    const int o   = blockIdx.x;
    const int tid = threadIdx.x;
    const size_t base = (size_t)o * IN_DIM;

    float swc = 0.0f, scc = 0.0f;
    for (int e = tid; e < IN_DIM; e += 256) {
        float w = W[base + e];
        float c = C[base + e];
        wb[base + e] = f2bf(w);
        cb[base + e] = f2bf(c);
        swc += w * c;
        scc += c * c;
    }
    r1[tid] = swc;
    r2[tid] = scc;
    __syncthreads();
    for (int s = 128; s > 0; s >>= 1) {
        if (tid < s) { r1[tid] += r1[tid + s]; r2[tid] += r2[tid + s]; }
        __syncthreads();
    }
    if (tid == 0) {
        bias[o] = -r1[0];
        csq[o]  =  r2[0];
        float sg = sigs[o];
        isig[o] = 1.0f / (sg * sg);
    }
}

// ---------------------------------------------------------------------------
// Pre-pass: convert x to bf16; xsq[b] = sum(x*x) in fp32.
// ---------------------------------------------------------------------------
__global__ __launch_bounds__(256)
void prep_x(const float* __restrict__ X,
            unsigned short* __restrict__ xb, float* __restrict__ xsq)
{
    __shared__ float r1[256];
    const int b   = blockIdx.x;
    const int tid = threadIdx.x;
    const size_t base = (size_t)b * IN_DIM;

    float s = 0.0f;
    for (int e = tid; e < IN_DIM; e += 256) {
        float x = X[base + e];
        xb[base + e] = f2bf(x);
        s += x * x;
    }
    r1[tid] = s;
    __syncthreads();
    for (int st = 128; st > 0; st >>= 1) {
        if (tid < st) r1[tid] += r1[tid + st];
        __syncthreads();
    }
    if (tid == 0) xsq[b] = r1[0];
}

// ---------------------------------------------------------------------------
// Stage one ROWS x 32 bf16 tile (64B/row) from global into LDS.
// 4 threads cover one row (16B chunks), coalesced.
// Async path: GLOBAL_LOAD_ASYNC_TO_LDS_B128 (ASYNCcnt), no VGPR roundtrip.
// Generic pointers: HIP's Default AS converts implicitly to the builtin's
// __device__/__shared__ parameter address spaces.
// ---------------------------------------------------------------------------
template <int ROWS>
__device__ inline void stage_tile(const unsigned short* __restrict__ g,
                                  int row0, int k0,
                                  unsigned char* lds, int tid)
{
    #pragma unroll
    for (int p = 0; p < ROWS / 64; ++p) {
        int c   = tid + 256 * p;
        int row = c >> 2;
        int ch  = c & 3;
        const unsigned short* src = g + (size_t)(row0 + row) * IN_DIM + k0 + ch * 8;
        unsigned char* dst = lds + row * LDS_ROW + ch * 16;
#if HAVE_ASYNC
        __builtin_amdgcn_global_load_async_to_lds_b128((v4i*)src, (v4i*)dst, 0, 0);
#else
        *reinterpret_cast<uint4*>(dst) = *reinterpret_cast<const uint4*>(src);
#endif
    }
}

// A fragment (16x32 bf16): lane = (m = lane&15, hi = lane>>4).
// Lane holds K = hi*16 + 0..15 contiguous (32B).
__device__ inline v16bf load_a_frag(const unsigned char* sA, int row0, int lm, int lhi)
{
    Frag f;
    const unsigned char* base = sA + (row0 + lm) * LDS_ROW + lhi * 32;
    f.q[0] = *reinterpret_cast<const uint4*>(base);
    f.q[1] = *reinterpret_cast<const uint4*>(base + 16);
    return f.v;
}

// B fragment (32x16 bf16): lane = (n = lane&15, hi = lane>>4).
// Lane holds K = hi*16 + 0..15 contiguous (32B).
__device__ inline v16bf load_b_frag(const unsigned char* sB, int col0, int lm, int lhi)
{
    Frag f;
    const unsigned char* base = sB + (col0 + lm) * LDS_ROW + lhi * 32;
    f.q[0] = *reinterpret_cast<const uint4*>(base);
    f.q[1] = *reinterpret_cast<const uint4*>(base + 16);
    return f.v;
}

// ---------------------------------------------------------------------------
// Main fused kernel: dual bf16 GEMM (x@W^T and x@C^T) + Gaussian epilogue.
// Grid: (OUT/BN, B/BM), block 256 threads = 8 wave32 waves (2 row x 4 col).
// ---------------------------------------------------------------------------
__global__ __launch_bounds__(256, 1)
void fgn_main(const unsigned short* __restrict__ xb,
              const unsigned short* __restrict__ wb,
              const unsigned short* __restrict__ cb,
              const float* __restrict__ bias,
              const float* __restrict__ csq,
              const float* __restrict__ isig,
              const float* __restrict__ xsq,
              float* __restrict__ out)
{
    extern __shared__ __align__(16) unsigned char smem[];

    const int tid  = threadIdx.x;
    const int lane = tid & 31;
    const int wave = tid >> 5;
    const int wr   = wave >> 2;   // 0..1 -> 64 rows each
    const int wc   = wave & 3;    // 0..3 -> 64 cols each
    const int lm   = lane & 15;
    const int lhi  = lane >> 4;

    const int bm0 = blockIdx.y * BM;
    const int bn0 = blockIdx.x * BN;

    v8f accW[4][4];
    v8f accC[4][4];
    #pragma unroll
    for (int i = 0; i < 4; ++i)
        #pragma unroll
        for (int j = 0; j < 4; ++j) { accW[i][j] = {}; accC[i][j] = {}; }

    // Prologue: stage k-tile 0 into buffer 0
    stage_tile<128>(xb, bm0, 0, smem, tid);
    stage_tile<256>(wb, bn0, 0, smem + TILE_A_BYTES, tid);
    stage_tile<256>(cb, bn0, 0, smem + TILE_A_BYTES + TILE_B_BYTES, tid);
    WAIT_ASYNC();
    __syncthreads();

    const int NK = IN_DIM / BK;   // 64
    for (int kt = 0; kt < NK; ++kt) {
        // Issue async stage of the next k-tile into the other buffer.
        // That buffer was last read in iteration kt-1, whose trailing barrier
        // already retired, so direct-to-LDS writes are safe.
        if (kt + 1 < NK) {
            int k0 = (kt + 1) * BK;
            unsigned char* nb = smem + ((kt + 1) & 1) * STAGE_BYTES;
            stage_tile<128>(xb, bm0, k0, nb, tid);
            stage_tile<256>(wb, bn0, k0, nb + TILE_A_BYTES, tid);
            stage_tile<256>(cb, bn0, k0, nb + TILE_A_BYTES + TILE_B_BYTES, tid);
        }

        const unsigned char* sA = smem + (kt & 1) * STAGE_BYTES;
        const unsigned char* sW = sA + TILE_A_BYTES;
        const unsigned char* sC = sA + TILE_A_BYTES + TILE_B_BYTES;

        v16bf af[4], bfw[4], bfc[4];
        #pragma unroll
        for (int i = 0; i < 4; ++i)
            af[i] = load_a_frag(sA, wr * 64 + i * 16, lm, lhi);
        #pragma unroll
        for (int j = 0; j < 4; ++j) {
            bfw[j] = load_b_frag(sW, wc * 64 + j * 16, lm, lhi);
            bfc[j] = load_b_frag(sC, wc * 64 + j * 16, lm, lhi);
        }

        #pragma unroll
        for (int i = 0; i < 4; ++i)
            #pragma unroll
            for (int j = 0; j < 4; ++j) {
                accW[i][j] = __builtin_amdgcn_wmma_f32_16x16x32_bf16(
                    false, af[i], false, bfw[j], (short)0, accW[i][j], false, false);
                accC[i][j] = __builtin_amdgcn_wmma_f32_16x16x32_bf16(
                    false, af[i], false, bfc[j], (short)0, accC[i][j], false, false);
            }

        // Commit: all of this wave's async loads done, then block-wide barrier.
        if (kt + 1 < NK) {
            WAIT_ASYNC();
            __syncthreads();
        }
    }

    // Fused epilogue: out = (xW + bias) * exp(-max(x2 + c2 - 2*xC, 0) / sig^2)
    #pragma unroll
    for (int i = 0; i < 4; ++i) {
        const int mbase = bm0 + wr * 64 + i * 16 + lhi * 8;
        float xs[8];
        #pragma unroll
        for (int v = 0; v < 8; ++v) xs[v] = xsq[mbase + v];

        #pragma unroll
        for (int j = 0; j < 4; ++j) {
            const int o  = bn0 + wc * 64 + j * 16 + lm;
            const float bo = bias[o];
            const float co = csq[o];
            const float io = isig[o];
            #pragma unroll
            for (int v = 0; v < 8; ++v) {
                const int brow = mbase + v;
                float l  = accW[i][j][v] + bo;
                float d2 = xs[v] + co - 2.0f * accC[i][j][v];
                d2 = fmaxf(d2, 0.0f);
                float g = __expf(-d2 * io);
                out[(size_t)brow * OUT_DIM + o] = l * g;
            }
        }
    }
}

// ---------------------------------------------------------------------------
// Launcher
// ---------------------------------------------------------------------------
extern "C" void kernel_launch(void* const* d_in, const int* in_sizes, int n_in,
                              void* d_out, int out_size, void* d_ws, size_t ws_size,
                              hipStream_t stream)
{
    const float* x    = (const float*)d_in[0];
    const float* W    = (const float*)d_in[1];
    const float* C    = (const float*)d_in[2];
    const float* sigs = (const float*)d_in[3];
    float* out        = (float*)d_out;

    unsigned char* ws = (unsigned char*)d_ws;
    size_t off = 0;
    auto alloc = [&](size_t bytes) -> void* {
        void* p = ws + off;
        off += (bytes + 255) & ~(size_t)255;
        return p;
    };
    unsigned short* xb = (unsigned short*)alloc((size_t)B_DIM * IN_DIM * 2);
    unsigned short* wb = (unsigned short*)alloc((size_t)OUT_DIM * IN_DIM * 2);
    unsigned short* cb = (unsigned short*)alloc((size_t)OUT_DIM * IN_DIM * 2);
    float* bias = (float*)alloc((size_t)OUT_DIM * 4);
    float* csq  = (float*)alloc((size_t)OUT_DIM * 4);
    float* isig = (float*)alloc((size_t)OUT_DIM * 4);
    float* xsq  = (float*)alloc((size_t)B_DIM * 4);

    prep_wc<<<OUT_DIM, 256, 0, stream>>>(W, C, sigs, wb, cb, bias, csq, isig);
    prep_x<<<B_DIM, 256, 0, stream>>>(x, xb, xsq);

    dim3 grid(OUT_DIM / BN, B_DIM / BM);
    fgn_main<<<grid, 256, SMEM_BYTES, stream>>>(xb, wb, cb, bias, csq, isig, xsq, out);
}
